// LinearAttentionFusedRecurrentKernel_57784490001093
// MI455X (gfx1250) — matrix operand order repaired and checked by reading, not compiled, not run on hardware
//
#include <hip/hip_runtime.h>

// Chunked linear attention for MI455X (gfx1250, wave32, WMMA).
// Grid: B*H*(D/16) = 128 workgroups; block = 128 threads = 4 waves.
// Each workgroup scans 64 chunks of 64 timesteps for one (b,h, v-slice).

typedef __attribute__((ext_vector_type(16))) __bf16 v16bf;
typedef __attribute__((ext_vector_type(8)))  float  v8f;

#define CHUNK 64
#define DM    64
#define SEQ   4096
#define NCHUNK (SEQ / CHUNK)

__device__ __forceinline__ v8f wmma_bf16(v16bf a, v16bf b, v8f c) {
    return __builtin_amdgcn_wmma_f32_16x16x32_bf16(
        /*neg_a=*/false, a, /*neg_b=*/false, b,
        /*c_mod=*/(short)0, c, /*reuse_a=*/false, /*reuse_b=*/false);
}

// A-matrix fragment (16x32 bf16): lane holds row m = lane%16.
// VGPR K layout per ISA 7.12.2: lanes 0-15 -> K 0..7 then 16..23;
// lanes 16-31 -> K 8..15 then 24..31.  Two 16B LDS reads.
__device__ __forceinline__ v16bf load_fragA(const __bf16* row, int lhalf) {
    union { v16bf v; uint4 q[2]; } f;
    f.q[0] = *(const uint4*)(row + lhalf * 8);
    f.q[1] = *(const uint4*)(row + 16 + lhalf * 8);
    return f.v;
}

// B-matrix fragment (32x16 bf16): lane holds column n = lane%16 of B,
// i.e. row n of B^T; lanes 0-15 hold K=0..7,16..23, lanes 16-31 hold K=8..15,24..31.
__device__ __forceinline__ v16bf load_fragB(const __bf16* rowT, int lhalf) {
    union { v16bf v; uint4 q[2]; } f;
    f.q[0] = *(const uint4*)(rowT + lhalf * 8);
    f.q[1] = *(const uint4*)(rowT + 16 + lhalf * 8);
    return f.v;
}

__global__ __launch_bounds__(128)
void linattn_chunked_wmma(const float* __restrict__ q,
                          const float* __restrict__ k,
                          const float* __restrict__ v,
                          float* __restrict__ out) {
    // LDS staging (bf16): 8+8+8+2+2+8 = 36 KB
    __shared__ __bf16 Qrm[CHUNK][DM];   // Q/8 row-major        (A of Q@H, Q@K^T)
    __shared__ __bf16 Krm[CHUNK][DM];   // K row-major = (K^T)^T (B of Q@K^T)
    __shared__ __bf16 KT [DM][CHUNK];   // K^T row-major         (A of K^T@V)
    __shared__ __bf16 VsT[16][CHUNK];   // V-slice^T             (B of S@V, K^T@V)
    __shared__ __bf16 HT [16][DM];      // H-slice^T snapshot    (B of Q@H)
    __shared__ __bf16 Sb [4][16][CHUNK];// masked S per wave     (A of S@V)

    const int bh    = blockIdx.x >> 2;   // 0..31  -> (b*H + h)
    const int vs    = blockIdx.x & 3;    // v-slice 0..3 (16 cols each)
    const int tid   = threadIdx.x;
    const int w     = tid >> 5;          // wave 0..3
    const int lane  = tid & 31;
    const int lhalf = lane >> 4;
    const int lmod  = lane & 15;

    const size_t seqbase = (size_t)bh * (size_t)(SEQ * DM);
    const float* qb = q + seqbase;
    const float* kb = k + seqbase;
    const float* vb = v + seqbase + vs * 16;
    float*       ob = out + seqbase + vs * 16;

    v8f h_acc = {};   // persistent f32 state tile: rows d in [16w,16w+16), cols = v-slice

    for (int c = 0; c < NCHUNK; ++c) {
        const float* qc = qb + (size_t)c * CHUNK * DM;
        const float* kc = kb + (size_t)c * CHUNK * DM;
        const float* vc = vb + (size_t)c * CHUNK * DM;

        // ---------- Phase 1: stage chunk into LDS as bf16 ----------
        #pragma unroll
        for (int i = 0; i < 8; ++i) {
            const int idx4 = i * 128 + tid;          // 1024 float4 = 64x64
            const int row  = idx4 >> 4;              // 16 float4 per row
            const int col  = (idx4 & 15) << 2;
            const float4 qv = ((const float4*)qc)[idx4];
            const float4 kv = ((const float4*)kc)[idx4];
            Qrm[row][col + 0] = (__bf16)(qv.x * 0.125f);
            Qrm[row][col + 1] = (__bf16)(qv.y * 0.125f);
            Qrm[row][col + 2] = (__bf16)(qv.z * 0.125f);
            Qrm[row][col + 3] = (__bf16)(qv.w * 0.125f);
            Krm[row][col + 0] = (__bf16)kv.x;
            Krm[row][col + 1] = (__bf16)kv.y;
            Krm[row][col + 2] = (__bf16)kv.z;
            Krm[row][col + 3] = (__bf16)kv.w;
            KT[col + 0][row] = (__bf16)kv.x;
            KT[col + 1][row] = (__bf16)kv.y;
            KT[col + 2][row] = (__bf16)kv.z;
            KT[col + 3][row] = (__bf16)kv.w;
        }
        #pragma unroll
        for (int i = 0; i < 2; ++i) {
            const int idx4 = i * 128 + tid;          // 256 float4 = 64x16 slice
            const int row  = idx4 >> 2;              // timestep in chunk
            const int col  = (idx4 & 3) << 2;        // v within slice
            const float4 vv = *(const float4*)(vc + (size_t)row * DM + col);
            VsT[col + 0][row] = (__bf16)vv.x;
            VsT[col + 1][row] = (__bf16)vv.y;
            VsT[col + 2][row] = (__bf16)vv.z;
            VsT[col + 3][row] = (__bf16)vv.w;
        }
        // Snapshot H^T (pre-update state) for the inter-chunk term.
        // C-layout: lanes 0-15 -> M=r, lanes 16-31 -> M=r+8; N = lmod.
        #pragma unroll
        for (int r = 0; r < 8; ++r)
            HT[lmod][w * 16 + lhalf * 8 + r] = (__bf16)h_acc[r];

        // Prefetch next chunk (global_prefetch_b8).
        if (c + 1 < NCHUNK) {
            __builtin_prefetch((const char*)(qc + CHUNK * DM) + tid * 128, 0, 1);
            __builtin_prefetch((const char*)(kc + CHUNK * DM) + tid * 128, 0, 1);
            if (tid < 64)
                __builtin_prefetch((const char*)(vc + CHUNK * DM) + (size_t)tid * DM * 4, 0, 1);
        }

        __syncthreads();

        // ---------- Phase 2: WMMA compute ----------
        const __bf16* qrow = &Qrm[w * 16 + lmod][0];
        const v16bf qf0 = load_fragA(qrow,      lhalf);
        const v16bf qf1 = load_fragA(qrow + 32, lhalf);

        // o = (Q/8) @ H_prev   (K-dim = d = 64 -> 2 steps)
        const __bf16* hrow = &HT[lmod][0];
        v8f o_acc = {};
        o_acc = wmma_bf16(qf0, load_fragB(hrow,      lhalf), o_acc);
        o_acc = wmma_bf16(qf1, load_fragB(hrow + 32, lhalf), o_acc);

        // S = (Q/8) @ K^T   (4 N-tiles x 2 K-steps)
        v8f s_acc[4];
        #pragma unroll
        for (int nt = 0; nt < 4; ++nt) {
            const __bf16* krow = &Krm[nt * 16 + lmod][0];
            v8f s = {};
            s = wmma_bf16(qf0, load_fragB(krow,      lhalf), s);
            s = wmma_bf16(qf1, load_fragB(krow + 32, lhalf), s);
            s_acc[nt] = s;
        }

        // Inclusive causal mask (n <= m), convert to bf16, park in LDS
        // to relayout C-format -> A-format.
        #pragma unroll
        for (int nt = 0; nt < 4; ++nt) {
            #pragma unroll
            for (int r = 0; r < 8; ++r) {
                const int m_loc = lhalf * 8 + r;            // row within wave tile
                const int n     = nt * 16 + lmod;           // column = timestep
                const float val = (n <= w * 16 + m_loc) ? s_acc[nt][r] : 0.0f;
                Sb[w][m_loc][n] = (__bf16)val;
            }
        }

        // o += S_masked @ V_slice   (K-dim = chunk time = 64)
        const __bf16* srow = &Sb[w][lmod][0];
        const __bf16* vrow = &VsT[lmod][0];
        const v16bf vf0 = load_fragB(vrow,      lhalf);
        const v16bf vf1 = load_fragB(vrow + 32, lhalf);
        o_acc = wmma_bf16(load_fragA(srow,      lhalf), vf0, o_acc);
        o_acc = wmma_bf16(load_fragA(srow + 32, lhalf), vf1, o_acc);

        // H += K^T @ V_slice  (f32 accumulator persists across chunks)
        const __bf16* ktrow = &KT[w * 16 + lmod][0];
        h_acc = wmma_bf16(load_fragA(ktrow,      lhalf), vf0, h_acc);
        h_acc = wmma_bf16(load_fragA(ktrow + 32, lhalf), vf1, h_acc);

        // Write output tile: row m = c*64 + w*16 + lhalf*8 + r, col = vs*16 + lmod
        float* orow = ob + (size_t)(c * CHUNK + w * 16 + lhalf * 8) * DM + lmod;
        #pragma unroll
        for (int r = 0; r < 8; ++r)
            orow[(size_t)r * DM] = o_acc[r];

        __syncthreads();   // protect LDS before next chunk's staging
    }
}

extern "C" void kernel_launch(void* const* d_in, const int* in_sizes, int n_in,
                              void* d_out, int out_size, void* d_ws, size_t ws_size,
                              hipStream_t stream) {
    (void)in_sizes; (void)n_in; (void)out_size; (void)d_ws; (void)ws_size;
    const float* q = (const float*)d_in[0];
    const float* k = (const float*)d_in[1];
    const float* v = (const float*)d_in[2];
    float* out = (float*)d_out;
    dim3 grid(2 * 16 * 4);   // B*H*(D/16) = 128 workgroups
    dim3 block(128);         // 4 waves of 32
    linattn_chunked_wmma<<<grid, block, 0, stream>>>(q, k, v, out);
}
